// ProtonXRNN_7473243095128
// MI455X (gfx1250) — matrix operand order repaired and checked by reading, not compiled, not run on hardware
//
#include <hip/hip_runtime.h>

// ---------------- problem constants ----------------
#define Bb   128
#define Tt   512
#define Ee   512
#define Uu   1024
#define N4   4096          // 4*U
#define NKT  48            // K / 32 (K = E + U = 1536)
#define NTN  256           // N4 / 16
#define NWG  64            // persistent workgroups; each owns 16 hidden units
#define BLK  256           // 8 waves

typedef __attribute__((ext_vector_type(16))) __bf16 v16bf;
typedef __attribute__((ext_vector_type(8)))  float  v8f;
typedef __attribute__((ext_vector_type(4)))  int    v4i;

#define AS1 __attribute__((address_space(1)))
#define AS3 __attribute__((address_space(3)))

typedef AS1 v4i* gv4i_p;   // pointer to int4 in global address space
typedef AS3 v4i* lv4i_p;   // pointer to int4 in LDS address space

__device__ __forceinline__ unsigned short f32_to_bf16(float f) {
    unsigned int u = __float_as_uint(f);
    unsigned int r = u + 0x7FFFu + ((u >> 16) & 1u);   // round-to-nearest-even
    return (unsigned short)(r >> 16);
}

__device__ __forceinline__ float sigmoidf_(float x) {
    return 1.0f / (1.0f + __expf(-x));
}

// ---- CDNA5 async global->LDS copy of 32 B (two b128 lanes), with fallback --
__device__ __forceinline__ void stage32B(const unsigned short* src,
                                         unsigned short* dst) {
#if __has_builtin(__builtin_amdgcn_global_load_async_to_lds_b128)
    __builtin_amdgcn_global_load_async_to_lds_b128(
        (gv4i_p)(src), (lv4i_p)(dst), 0, 0);
    __builtin_amdgcn_global_load_async_to_lds_b128(
        (gv4i_p)(src + 8), (lv4i_p)(dst + 8), 0, 0);
#else
    const uint4* s4 = (const uint4*)src;
    uint4* d4 = (uint4*)dst;
    d4[0] = s4[0];
    d4[1] = s4[1];
#endif
}

__device__ __forceinline__ void wait_async0() {
#if __has_builtin(__builtin_amdgcn_s_wait_asynccnt)
    __builtin_amdgcn_s_wait_asynccnt(0);
#else
    asm volatile("s_wait_asynccnt 0x0" ::: "memory");
#endif
}

// ---------------------------------------------------------------------------
// Pack [Wx; Wh] (fp32, row-major (K,N)) into bf16 WMMA B-fragment layout:
// tile (kt, nt) covers K rows [kt*32, kt*32+32), N cols [nt*16, nt*16+16).
// Lane l holds 16 contiguous K values for column n = nt*16 + l%16,
// k base = kt*32 + (l<16 ? 0 : 16).  Stored contiguously: 32 B per lane.
// ---------------------------------------------------------------------------
__global__ __launch_bounds__(BLK) void pack_w_kernel(
    const float* __restrict__ Wx, const float* __restrict__ Wh,
    unsigned short* __restrict__ Wpack)
{
    int g = blockIdx.x * blockDim.x + threadIdx.x;     // tile*32 + lane
    if (g >= NKT * NTN * 32) return;
    int tile = g >> 5;
    int lane = g & 31;
    int kt = tile >> 8;
    int nt = tile & 255;
    int n  = nt * 16 + (lane & 15);
    int kb = kt * 32 + ((lane < 16) ? 0 : 16);
    unsigned short* dst = Wpack + (size_t)g * 16;
#pragma unroll
    for (int j = 0; j < 16; ++j) {
        int k = kb + j;
        float w = (k < Ee) ? Wx[(size_t)k * N4 + n]
                           : Wh[(size_t)(k - Ee) * N4 + n];
        dst[j] = f32_to_bf16(w);
    }
}

// ---------------- grid-wide barrier (monotonic counter) --------------------
__device__ __forceinline__ void grid_barrier(int* bar, int phase) {
    __syncthreads();
    if (threadIdx.x == 0) {
        __threadfence();
        __hip_atomic_fetch_add(bar, 1, __ATOMIC_ACQ_REL, __HIP_MEMORY_SCOPE_AGENT);
        const int target = NWG * (phase + 1);
        while (__hip_atomic_load(bar, __ATOMIC_ACQUIRE, __HIP_MEMORY_SCOPE_AGENT) < target) {
            __builtin_amdgcn_s_sleep(1);
        }
    }
    __syncthreads();
}

// ---------------------------------------------------------------------------
// One GEMM phase: NSLAB K-slabs of 32 from `base` (row stride `stride`),
// double-buffered A slab in LDS via async loads, 4 WMMAs per wave per slab.
// ---------------------------------------------------------------------------
template <int NSLAB>
__device__ __forceinline__ void gemm_phase(
    const unsigned short* __restrict__ base, int stride, int ktGlobalBase,
    unsigned short* A0, unsigned short* A1,
    const unsigned short* __restrict__ Wpack, int ntile,
    int tid, int lane, int mgrp, v8f (&acc)[4])
{
    const int r    = tid >> 1;
    const int half = tid & 1;
    const int arow = lane & 15;
    const int kb   = (lane < 16) ? 0 : 16;

    // prologue: stage slab 0
    stage32B(base + (size_t)r * stride + half * 16, A0 + r * 32 + half * 16);
    wait_async0();
    __syncthreads();

#pragma unroll 1
    for (int kt = 0; kt < NSLAB; ++kt) {
        unsigned short* cbuf = (kt & 1) ? A1 : A0;
        unsigned short* nbuf = (kt & 1) ? A0 : A1;
        if (kt + 1 < NSLAB) {
            stage32B(base + (size_t)r * stride + (kt + 1) * 32 + half * 16,
                     nbuf + r * 32 + half * 16);
        }

        // B fragment: one contiguous 32 B load per lane (L2-resident)
        union { uint4 q[2]; v16bf v; } bf_;
        const uint4* bsrc = (const uint4*)(Wpack +
            ((size_t)((ktGlobalBase + kt) * NTN + ntile) * 32 + lane) * 16);
        bf_.q[0] = bsrc[0];
        bf_.q[1] = bsrc[1];
        if (kt + 1 < NSLAB) {
            __builtin_prefetch((const void*)(Wpack +
                ((size_t)((ktGlobalBase + kt + 1) * NTN + ntile) * 32 + lane) * 16),
                0, 3);
        }

#pragma unroll
        for (int m = 0; m < 4; ++m) {
            int rr = (mgrp * 4 + m) * 16 + arow;
            union { uint4 q[2]; v16bf v; } af_;
            af_.q[0] = *(const uint4*)(cbuf + rr * 32 + kb);
            af_.q[1] = *(const uint4*)(cbuf + rr * 32 + kb + 8);
            acc[m] = __builtin_amdgcn_wmma_f32_16x16x32_bf16(
                false, af_.v, false, bf_.v, (short)0, acc[m], false, false);
        }

        wait_async0();      // our async writes to nbuf landed
        __syncthreads();    // everyone's writes visible; cbuf reads done
    }
}

// ---------------------------------------------------------------------------
// Persistent fused LSTM.  Per step t:
//   z = [x_t | h_{t-1}] (bf16, 128x1536) @ Wpack (bf16, 1536x4096) + bias
//   gates -> c, h ;  h_t (bf16) -> hbuf[next];  gather x_{t+1} -> xbuf[next]
// ---------------------------------------------------------------------------
__global__ __launch_bounds__(BLK) void lstm_persistent_kernel(
    const int*   __restrict__ sentence,     // (B, T)
    const float* __restrict__ emb,          // (V, E)
    const float* __restrict__ bias,         // (4U)
    const unsigned short* __restrict__ Wpack,
    unsigned short* __restrict__ xbuf,      // 2 * B * E  bf16
    unsigned short* __restrict__ hbuf,      // 2 * B * U  bf16
    float* __restrict__ hfinal,             // B * U  fp32
    int* __restrict__ bar)
{
    __shared__ __align__(16) unsigned short A0[128 * 32];      // 8 KB slab buf
    __shared__ __align__(16) unsigned short A1[128 * 32];      // 8 KB slab buf
    __shared__ __align__(16) float z_lds[128 * 64];            // 32 KB z slice
    __shared__ __align__(16) float c_lds[128 * 16];            // 8 KB cell state
    __shared__ float bias_lds[64];

    const int tid  = threadIdx.x;
    const int wg   = blockIdx.x;          // 0..63
    const int wave = tid >> 5;            // 0..7
    const int lane = tid & 31;
    const int gate = wave & 3;            // 0..3
    const int mgrp = wave >> 2;           // 0..1
    const int u0   = wg * 16;
    const int ntile = gate * (Uu / 16) + wg;

    // ---------------- init -------------------------------------------------
    for (int i = tid; i < 128 * 16; i += BLK) c_lds[i] = 0.0f;
    if (tid < 64) bias_lds[tid] = bias[(tid >> 4) * Uu + u0 + (tid & 15)];
    for (int rr = 0; rr < 2; ++rr) {
        int b = wg * 2 + rr;
        int tok = sentence[b * Tt + 0];
        const float* src = emb + (size_t)tok * Ee;
        for (int e = tid; e < Ee; e += BLK)
            xbuf[b * Ee + e] = f32_to_bf16(src[e]);
    }
    for (int i = tid; i < 128 * 16; i += BLK) {
        int r = i >> 4, uu = i & 15;
        hbuf[r * Uu + u0 + uu] = 0;
    }
    grid_barrier(bar, 0);

    // ---------------- time loop -------------------------------------------
    for (int t = 0; t < Tt; ++t) {
        const int cur = t & 1, nxt = cur ^ 1;
        const unsigned short* xb = xbuf + (size_t)cur * (Bb * Ee);
        const unsigned short* hb = hbuf + (size_t)cur * (Bb * Uu);

        v8f acc[4] = {};
        gemm_phase<16>(xb, Ee,  0, A0, A1, Wpack, ntile, tid, lane, mgrp, acc);
        gemm_phase<32>(hb, Uu, 16, A0, A1, Wpack, ntile, tid, lane, mgrp, acc);

        // scatter accumulators to z_lds (C/D layout: VGPR v -> M = v + 8*(lane>=16))
#pragma unroll
        for (int m = 0; m < 4; ++m) {
            int rbase = (mgrp * 4 + m) * 16 + ((lane >= 16) ? 8 : 0);
            int col   = gate * 16 + (lane & 15);
#pragma unroll
            for (int v = 0; v < 8; ++v)
                z_lds[(rbase + v) * 64 + col] = acc[m][v];
        }
        __syncthreads();

        // gates + cell update for this WG's 128x16 slice
        for (int i = tid; i < 128 * 16; i += BLK) {
            int r = i >> 4, uu = i & 15;
            float zi = z_lds[r * 64 +      uu] + bias_lds[     uu];
            float zf = z_lds[r * 64 + 16 + uu] + bias_lds[16 + uu];
            float zg = z_lds[r * 64 + 32 + uu] + bias_lds[32 + uu];
            float zo = z_lds[r * 64 + 48 + uu] + bias_lds[48 + uu];
            float ig = sigmoidf_(zi);
            float fg = sigmoidf_(zf);
            float gg = tanhf(zg);
            float og = sigmoidf_(zo);
            float c  = fg * c_lds[i] + ig * gg;
            c_lds[i] = c;
            float h  = og * tanhf(c);
            hbuf[(size_t)nxt * (Bb * Uu) + r * Uu + u0 + uu] = f32_to_bf16(h);
            if (t == Tt - 1) hfinal[r * Uu + u0 + uu] = h;
        }

        // gather x_{t+1} for this WG's two batch rows
        if (t + 1 < Tt) {
            for (int rr = 0; rr < 2; ++rr) {
                int b = wg * 2 + rr;
                int tok = sentence[b * Tt + t + 1];
                const float* src = emb + (size_t)tok * Ee;
                unsigned short* dst = xbuf + (size_t)nxt * (Bb * Ee) + b * Ee;
                for (int e = tid; e < Ee; e += BLK)
                    dst[e] = f32_to_bf16(src[e]);
            }
        }
        grid_barrier(bar, t + 1);
    }
}

// ---------------- tiny MLP head: sigmoid((h@W1+b1)@W2+b2) ------------------
__global__ __launch_bounds__(64) void head_kernel(
    const float* __restrict__ hfinal, const float* __restrict__ W1,
    const float* __restrict__ b1, const float* __restrict__ W2,
    const float* __restrict__ b2, float* __restrict__ out)
{
    __shared__ float s[64];
    int b = blockIdx.x, j = threadIdx.x;
    const float* hr = hfinal + (size_t)b * Uu;
    float acc = b1[j];
    for (int u = 0; u < Uu; ++u) acc += hr[u] * W1[(size_t)u * 64 + j];
    s[j] = acc * W2[j];
    __syncthreads();
    for (int off = 32; off > 0; off >>= 1) {
        if (j < off) s[j] += s[j + off];
        __syncthreads();
    }
    if (j == 0) out[b] = 1.0f / (1.0f + __expf(-(s[0] + b2[0])));
}

// ---------------------------------------------------------------------------
extern "C" void kernel_launch(void* const* d_in, const int* in_sizes, int n_in,
                              void* d_out, int out_size, void* d_ws, size_t ws_size,
                              hipStream_t stream) {
    const int*   sentence = (const int*)  d_in[0];
    const float* emb      = (const float*)d_in[1];
    const float* Wx       = (const float*)d_in[2];
    const float* Wh       = (const float*)d_in[3];
    const float* bias     = (const float*)d_in[4];
    const float* W1       = (const float*)d_in[5];
    const float* b1       = (const float*)d_in[6];
    const float* W2       = (const float*)d_in[7];
    const float* b2       = (const float*)d_in[8];
    float* out = (float*)d_out;

    char* ws = (char*)d_ws;
    size_t off = 0;
    int* bar = (int*)(ws + off);                    off += 256;
    unsigned short* Wpack = (unsigned short*)(ws + off);
    off += (size_t)NKT * NTN * 512 * sizeof(unsigned short);   // 12.58 MB
    unsigned short* xbuf = (unsigned short*)(ws + off);
    off += (size_t)2 * Bb * Ee * sizeof(unsigned short);
    unsigned short* hbuf = (unsigned short*)(ws + off);
    off += (size_t)2 * Bb * Uu * sizeof(unsigned short);
    float* hfinal = (float*)(ws + off);
    off += (size_t)Bb * Uu * sizeof(float);

    (void)hipMemsetAsync(bar, 0, 256, stream);

    {   // one-time weight pack (bf16, WMMA B layout)
        int total = NKT * NTN * 32;
        pack_w_kernel<<<(total + BLK - 1) / BLK, BLK, 0, stream>>>(Wx, Wh, Wpack);
    }

    lstm_persistent_kernel<<<NWG, BLK, 0, stream>>>(
        sentence, emb, bias, Wpack, xbuf, hbuf, hfinal, bar);

    head_kernel<<<Bb, 64, 0, stream>>>(hfinal, W1, b1, W2, b2, out);
}
